// DCN_MTMD_6227702579405
// MI455X (gfx1250) — matrix operand order repaired and checked by reading, not compiled, not run on hardware
//
#include <hip/hip_runtime.h>
#include <hip/hip_bf16.h>

// ---------------------------------------------------------------------------
// DCN + MMoE multi-domain model forward for MI455X (gfx1250, wave32, WMMA).
// GEMMs run on v_wmma_f32_16x16x32_bf16 (bf16 in, fp32 accumulate).
// LDS layouts are arranged so every WMMA fragment is built from aligned
// 16-byte ds_load_b128 ops (A rows padded to 80B; B stored transposed).
// ---------------------------------------------------------------------------

typedef __bf16  bf16;
typedef __attribute__((ext_vector_type(16))) __bf16 v16bf;
typedef __attribute__((ext_vector_type(8)))  __bf16 v8bf;
typedef __attribute__((ext_vector_type(8)))  float  v8f;

#define WMMA_BF16(a, b, c) \
  __builtin_amdgcn_wmma_f32_16x16x32_bf16(false, (a), false, (b), (short)0, (c), false, false)

static constexpr int   Bsz  = 16384;
static constexpr int   BC   = 2048;     // batch chunk for expert pipeline
static constexpr int   NCHK = Bsz / BC;
static constexpr float EPS  = 1e-5f;
static constexpr int   LPAD = 40;       // LDS row stride (bf16 elems): 80B, 16B-aligned chunks

// Build a 16-element bf16 fragment from two aligned 16-byte LDS chunks.
__device__ __forceinline__ v16bf ld_frag(const bf16* p0, const bf16* p1) {
  v8bf lo = *(const v8bf*)p0;
  v8bf hi = *(const v8bf*)p1;
  return __builtin_shufflevector(lo, hi, 0, 1, 2, 3, 4, 5, 6, 7,
                                         8, 9, 10, 11, 12, 13, 14, 15);
}

// ---------------------------------------------------------------------------
// fp32 -> bf16 conversion
// ---------------------------------------------------------------------------
__global__ __launch_bounds__(256) void cvt_bf16_kernel(const float* __restrict__ src,
                                                       bf16* __restrict__ dst, long n) {
  long i = (long)blockIdx.x * 256 + threadIdx.x;
  if (i < n) dst[i] = (bf16)src[i];
}

// out[g*C+c] = sum_r w[g*R*C + r*C + c]   (collapses rank-1 broadcast GEMMs)
__global__ __launch_bounds__(256) void colsum_kernel(const float* __restrict__ w,
                                                     float* __restrict__ out,
                                                     int G, int R, int C) {
  int i = blockIdx.x * 256 + threadIdx.x;
  if (i >= G * C) return;
  int g = i / C, c = i % C;
  const float* p = w + (long)g * R * C + c;
  float s = 0.f;
  for (int r = 0; r < R; ++r) s += p[(long)r * C];
  out[i] = s;
}

// ---------------------------------------------------------------------------
// Embedding gather + 3-layer DCN cross. One block per batch row, 256 threads.
// ---------------------------------------------------------------------------
__global__ __launch_bounds__(256) void embed_cross_kernel(
    const int* __restrict__ x, const float* __restrict__ emb,
    const float* __restrict__ cw, const float* __restrict__ cb,
    float* __restrict__ hout, bf16* __restrict__ exbf) {
  int b = blockIdx.x;
  int d = threadIdx.x;                 // 0..255 (F*D)
  int f = d >> 4;
  int idx = x[f * Bsz + b];
  float ex = emb[((long)f * 100000 + idx) * 16 + (d & 15)];
  exbf[(long)b * 256 + d] = (bf16)ex;

  __shared__ float red[256];
  float h = ex;
  for (int i = 0; i < 3; ++i) {
    red[d] = h * cw[i * 256 + d];
    __syncthreads();
    for (int s = 128; s > 0; s >>= 1) {
      if (d < s) red[d] += red[d + s];
      __syncthreads();
    }
    float dot = red[0];
    __syncthreads();
    h = ex * dot + cb[i * 256 + d] + h;
  }
  hout[(long)b * 256 + d] = h;
}

// ---------------------------------------------------------------------------
// Generic batched bf16 WMMA GEMM:  C(f32)[M,N] = A[M,K] @ W[K,N] + bias[N]
// Block = 256 threads (8 waves), tile 128x64, K-step 32.
// Wave grid 4x2, each wave computes a 32x32 tile (2x2 WMMA 16x16x32 tiles).
// A in LDS row-major (pad 40), B in LDS TRANSPOSED (col-major, pad 40) so all
// fragment reads are aligned ds_load_b128.
// ---------------------------------------------------------------------------
__global__ __launch_bounds__(256) void gemm_bias_kernel(
    const bf16* __restrict__ A, const bf16* __restrict__ W,
    const float* __restrict__ bias, float* __restrict__ C,
    int M, int N, int K,
    long strideA, long strideW, long strideBias, long strideC) {
  A    += (long)blockIdx.z * strideA;
  W    += (long)blockIdx.z * strideW;
  bias += (long)blockIdx.z * strideBias;
  C    += (long)blockIdx.z * strideC;

  __shared__ bf16 sA[128 * LPAD];      // [row][k]  (k-chunks 16B aligned)
  __shared__ bf16 sBT[64 * LPAD];      // [col][k]  (transposed B)

  const int tid   = threadIdx.x;
  const int w     = tid >> 5;
  const int lane  = tid & 31;
  const int lrow  = lane & 15;
  const int lhalf = lane >> 4;
  const int wrow  = w & 3;             // 4 row groups of 32 rows
  const int wcol  = w >> 2;            // 2 col groups of 32 cols
  const int m0    = blockIdx.x * 128;
  const int n0    = blockIdx.y * 64;

  v8f acc[2][2] = {};

  for (int k0 = 0; k0 < K; k0 += 32) {
    {  // stage A tile 128x32: each thread copies 16 contiguous bf16 (2 x b128)
      int r = tid >> 1, cs = (tid & 1) * 16;
      const v8bf* src = (const v8bf*)(A + (long)(m0 + r) * K + k0 + cs);
      v8bf* dst = (v8bf*)(sA + r * LPAD + cs);
      dst[0] = src[0];
      dst[1] = src[1];
    }
    {  // stage B tile 32x64 transposed: b128 global read, scatter b16 to LDS
      int k = tid >> 3, nseg = (tid & 7) * 8;
      v8bf vw = *(const v8bf*)(W + (long)(k0 + k) * N + n0 + nseg);
#pragma unroll
      for (int i = 0; i < 8; ++i) sBT[(nseg + i) * LPAD + k] = vw[i];
    }
    __syncthreads();

    v16bf af[2], bfr[2];
#pragma unroll
    for (int mi = 0; mi < 2; ++mi) {   // A: lanes0-15 K{0..7,16..23}, lanes16-31 +8
      const bf16* base = sA + (wrow * 32 + mi * 16 + lrow) * LPAD + lhalf * 8;
      af[mi] = ld_frag(base, base + 16);
    }
#pragma unroll
    for (int ni = 0; ni < 2; ++ni) {   // B: lanes0-15 K{0..7,16..23}, lanes16-31 +8
      const bf16* base = sBT + (wcol * 32 + ni * 16 + lrow) * LPAD + lhalf * 8;
      bfr[ni] = ld_frag(base, base + 16);
    }
#pragma unroll
    for (int mi = 0; mi < 2; ++mi)
#pragma unroll
      for (int ni = 0; ni < 2; ++ni)
        acc[mi][ni] = WMMA_BF16(af[mi], bfr[ni], acc[mi][ni]);
    __syncthreads();
  }

#pragma unroll
  for (int mi = 0; mi < 2; ++mi)
#pragma unroll
    for (int ni = 0; ni < 2; ++ni) {
      int colg = n0 + wcol * 32 + ni * 16 + lrow;
      float bv = bias ? bias[colg] : 0.f;
#pragma unroll
      for (int r = 0; r < 8; ++r) {    // C layout: M = (lane/16)*8 + r, N = lane%16
        int rowg = m0 + wrow * 32 + mi * 16 + lhalf * 8 + r;
        C[(long)rowg * N + colg] = acc[mi][ni][r] + bv;
      }
    }
}

// ---------------------------------------------------------------------------
// Batched (per-expert) bf16 WMMA GEMM with fused bias + LayerNorm + ReLU.
// Block covers the FULL row width N, BM=32 rows, so the LN row reduction
// happens in LDS. N is a template parameter -> NT-tile loop fully unrolled.
// Wave grid 2x4: each wave = 16 rows x N/4 cols (NT = N/64 WMMA col tiles).
// ---------------------------------------------------------------------------
template <int N>
__global__ __launch_bounds__(256) void gemm_ln_relu_kernel(
    const bf16* __restrict__ A, const bf16* __restrict__ W,
    const float* __restrict__ bias, const float* __restrict__ gam,
    const float* __restrict__ bet, bf16* __restrict__ Out,
    int K, long strideA, long strideW, long strideOut) {
  const int e = blockIdx.z;
  A += e * strideA; W += e * strideW; Out += e * strideOut;
  bias += (long)e * N; gam += (long)e * N; bet += (long)e * N;

  extern __shared__ char smem[];
  bf16*  sA  = (bf16*)smem;                                  // [32][LPAD]
  bf16*  sBT = (bf16*)(smem + 32 * LPAD * 2);                // [N][LPAD] transposed
  float* sC  = (float*)(smem + 32 * LPAD * 2 + N * LPAD * 2);// [32][N]
  __shared__ float rmean[32], rrstd[32];

  const int tid   = threadIdx.x;
  const int w     = tid >> 5;
  const int lane  = tid & 31;
  const int lrow  = lane & 15;
  const int lhalf = lane >> 4;
  const int wrow  = w & 1;            // 2 row groups of 16
  const int wcol  = w >> 1;           // 4 col groups of N/4
  constexpr int NT = N >> 6;          // WMMA col tiles per wave (4 or 8)
  const int m0    = blockIdx.x * 32;
  const int colbase = wcol * (N >> 2);

  v8f acc[NT] = {};

  for (int k0 = 0; k0 < K; k0 += 32) {
    if (tid < 64) {  // A 32x32: 2 x b128 per thread
      int r = tid >> 1, cs = (tid & 1) * 16;
      const v8bf* src = (const v8bf*)(A + (long)(m0 + r) * K + k0 + cs);
      v8bf* dst = (v8bf*)(sA + r * LPAD + cs);
      dst[0] = src[0];
      dst[1] = src[1];
    }
    {  // B 32xN transposed: N/64 b128 reads per thread, scatter b16
      int k = tid >> 3;
#pragma unroll
      for (int g = 0; g < NT; ++g) {
        int nseg = (tid & 7) * 8 + g * 64;
        v8bf vw = *(const v8bf*)(W + (long)(k0 + k) * N + nseg);
#pragma unroll
        for (int i = 0; i < 8; ++i) sBT[(nseg + i) * LPAD + k] = vw[i];
      }
    }
    __syncthreads();

    v16bf af;
    {
      const bf16* base = sA + (wrow * 16 + lrow) * LPAD + lhalf * 8;
      af = ld_frag(base, base + 16);
    }
#pragma unroll
    for (int ni = 0; ni < NT; ++ni) {
      const bf16* base = sBT + (colbase + ni * 16 + lrow) * LPAD + lhalf * 8;
      v16bf bfr = ld_frag(base, base + 16);
      acc[ni] = WMMA_BF16(af, bfr, acc[ni]);
    }
    __syncthreads();
  }

  // accum + bias -> LDS
#pragma unroll
  for (int ni = 0; ni < NT; ++ni) {
    int col = colbase + ni * 16 + lrow;
    float bv = bias[col];
#pragma unroll
    for (int r = 0; r < 8; ++r) {
      int row = wrow * 16 + lhalf * 8 + r;
      sC[row * N + col] = acc[ni][r] + bv;
    }
  }
  __syncthreads();

  if (tid < 32) {  // per-row LN stats
    const float* rowp = sC + tid * N;
    float s = 0.f, s2 = 0.f;
    for (int c = 0; c < N; ++c) { float v = rowp[c]; s += v; s2 += v * v; }
    float mn = s / N;
    float vr = s2 / N - mn * mn;
    rmean[tid] = mn;
    rrstd[tid] = rsqrtf(vr + EPS);
  }
  __syncthreads();

  {  // normalize + relu + store bf16 (8 threads per row)
    int row = tid >> 3;
    float mn = rmean[row], rs = rrstd[row];
    const float* rowp = sC + row * N;
    bf16* dst = Out + (long)(m0 + row) * N;
    for (int c = tid & 7; c < N; c += 8) {
      float v = (rowp[c] - mn) * rs * gam[c] + bet[c];
      dst[c] = (bf16)(v > 0.f ? v : 0.f);
    }
  }
}

// ---------------------------------------------------------------------------
// Batch-norm statistics over the batch axis: one block per column.
// ---------------------------------------------------------------------------
__global__ __launch_bounds__(256) void bn_stats_kernel(
    const float* __restrict__ z, int Bn, int N,
    float* __restrict__ mean, float* __restrict__ inv) {
  int n = blockIdx.x;
  float s = 0.f, s2 = 0.f;
  for (int r = threadIdx.x; r < Bn; r += 256) {
    float v = z[(long)r * N + n];
    s += v; s2 += v * v;
  }
  __shared__ float rs[256], rs2[256];
  rs[threadIdx.x] = s; rs2[threadIdx.x] = s2;
  __syncthreads();
  for (int st = 128; st > 0; st >>= 1) {
    if (threadIdx.x < st) { rs[threadIdx.x] += rs[threadIdx.x + st]; rs2[threadIdx.x] += rs2[threadIdx.x + st]; }
    __syncthreads();
  }
  if (threadIdx.x == 0) {
    float m = rs[0] / Bn;
    float v = rs2[0] / Bn - m * m;
    mean[n] = m;
    inv[n]  = rsqrtf(v + EPS);
  }
}

__global__ __launch_bounds__(256) void bn_relu_kernel(
    const float* __restrict__ z, const float* __restrict__ mean,
    const float* __restrict__ inv, const float* __restrict__ g,
    const float* __restrict__ be, bf16* __restrict__ outbf,
    float* __restrict__ outf, long total, int N) {
  long i = (long)blockIdx.x * 256 + threadIdx.x;
  if (i >= total) return;
  int n = (int)(i % N);
  float v = (z[i] - mean[n]) * inv[n] * g[n] + be[n];
  v = v > 0.f ? v : 0.f;
  if (outbf) outbf[i] = (bf16)v;
  if (outf)  outf[i]  = v;
}

// ---------------------------------------------------------------------------
// y[b] = concat(h, m) . lrw + lrb    (one block per row)
// ---------------------------------------------------------------------------
__global__ __launch_bounds__(256) void y_kernel(
    const float* __restrict__ h, const float* __restrict__ m,
    const float* __restrict__ lrw, const float* __restrict__ lrb,
    float* __restrict__ y) {
  int b = blockIdx.x, t = threadIdx.x;
  float v = h[(long)b * 256 + t] * lrw[t];
  if (t < 128) v += m[(long)b * 128 + t] * lrw[256 + t];
  __shared__ float red[256];
  red[t] = v;
  __syncthreads();
  for (int s = 128; s > 0; s >>= 1) {
    if (t < s) red[t] += red[t + s];
    __syncthreads();
  }
  if (t == 0) y[b] = red[0] + lrb[0];
}

// Gate softmax: logits collapse to y[b]*colsum(gw[g]) + gb[g].
__global__ __launch_bounds__(256) void gate_kernel(
    const float* __restrict__ y, const float* __restrict__ gwsum,
    const float* __restrict__ gb, float* __restrict__ gcoef) {
  int b = blockIdx.x * 256 + threadIdx.x;
  if (b >= Bsz) return;
  float yv = y[b];
  for (int g = 0; g < 6; ++g) {
    float lg[10], mx = -1e30f;
    for (int o = 0; o < 10; ++o) {
      lg[o] = yv * gwsum[g * 10 + o] + gb[g * 10 + o];
      mx = fmaxf(mx, lg[o]);
    }
    float s = 0.f;
    for (int o = 0; o < 10; ++o) { lg[o] = __expf(lg[o] - mx); s += lg[o]; }
    float rs = 1.f / s;
    float* dst = gcoef + ((long)g * Bsz + b) * 10;
    for (int o = 0; o < 10; ++o) dst[o] = lg[o] * rs;
  }
}

// Expert layer 0 (rank-1 collapse) + LN + ReLU. One block per (row, expert).
__global__ __launch_bounds__(256) void expert0_ln_kernel(
    const float* __restrict__ y, const float* __restrict__ ewsum0,
    const float* __restrict__ eb0, const float* __restrict__ eg0,
    const float* __restrict__ ebe0, bf16* __restrict__ heA, int b0) {
  int e = blockIdx.y, bl = blockIdx.x, t = threadIdx.x;
  float yv = y[b0 + bl];
  const float* s  = ewsum0 + e * 512;
  const float* bb = eb0 + e * 512;
  float v0 = yv * s[t] + bb[t];
  float v1 = yv * s[t + 256] + bb[t + 256];
  __shared__ float r1[256], r2[256];
  r1[t] = v0 + v1; r2[t] = v0 * v0 + v1 * v1;
  __syncthreads();
  for (int st = 128; st > 0; st >>= 1) {
    if (t < st) { r1[t] += r1[t + st]; r2[t] += r2[t + st]; }
    __syncthreads();
  }
  float mean = r1[0] / 512.f;
  float var  = r2[0] / 512.f - mean * mean;
  float rstd = rsqrtf(var + EPS);
  bf16* dst = heA + ((long)e * BC + bl) * 512;
  float o0 = (v0 - mean) * rstd * eg0[e * 512 + t]       + ebe0[e * 512 + t];
  float o1 = (v1 - mean) * rstd * eg0[e * 512 + t + 256] + ebe0[e * 512 + t + 256];
  dst[t]       = (bf16)(o0 > 0.f ? o0 : 0.f);
  dst[t + 256] = (bf16)(o1 > 0.f ? o1 : 0.f);
}

// Mixture fuse: shared experts + gated domain/task experts -> bf16 tower input.
__global__ __launch_bounds__(256) void fuse_kernel(
    const float* __restrict__ gcoef, const bf16* __restrict__ he3,
    const int* __restrict__ x, bf16* __restrict__ fused, int b0) {
  long i = (long)blockIdx.x * 256 + threadIdx.x;
  if (i >= (long)6 * BC * 256) return;
  int hcol = (int)(i & 255);
  long rem = i >> 8;
  int bl = (int)(rem % BC);
  int g  = (int)(rem / BC);
  int bg = b0 + bl;
  const float* gc = gcoef + ((long)g * Bsz + bg) * 10;
  float acc = 0.f;
  for (int e = 0; e < 8; ++e)
    acc += gc[e] * (float)he3[((long)e * BC + bl) * 256 + hcol];
  int d = g % 3, t = g % 2;
  int dom = x[15 * Bsz + bg];
  float dsel = (dom == d) ? (float)he3[((long)(8 + d) * BC + bl) * 256 + hcol] : 0.f;
  float tsel = (float)he3[((long)(11 + t) * BC + bl) * 256 + hcol];
  acc += gc[8] * dsel + gc[9] * tsel;
  fused[((long)g * BC + bl) * 256 + hcol] = (bf16)acc;
}

// Tower BN + ReLU + dot(tw2) + sigmoid, only for the 2 gates each row needs.
__global__ __launch_bounds__(256) void final_kernel(
    const float* __restrict__ t1z, const float* __restrict__ tmean,
    const float* __restrict__ tinv, const float* __restrict__ tg,
    const float* __restrict__ tbe, const float* __restrict__ tw2,
    const float* __restrict__ tb2, const int* __restrict__ x,
    float* __restrict__ out) {
  int b = blockIdx.x * 256 + threadIdx.x;
  if (b >= Bsz) return;
  int dom = x[15 * Bsz + b];
  for (int t = 0; t < 2; ++t) {
    int g = t * 3 + dom;
    const float* z = t1z + ((long)g * Bsz + b) * 256;
    float acc = 0.f;
    for (int h = 0; h < 256; ++h) {
      float v = (z[h] - tmean[g * 256 + h]) * tinv[g * 256 + h] * tg[g * 256 + h] + tbe[g * 256 + h];
      v = v > 0.f ? v : 0.f;
      acc += v * tw2[g * 256 + h];
    }
    out[b * 2 + t] = 1.f / (1.f + __expf(-(acc + tb2[g])));
  }
}

// ---------------------------------------------------------------------------
// Host orchestration
// ---------------------------------------------------------------------------
extern "C" void kernel_launch(void* const* d_in, const int* in_sizes, int n_in,
                              void* d_out, int out_size, void* d_ws, size_t ws_size,
                              hipStream_t stream) {
  const int*   x    = (const int*)d_in[0];
  const float* emb  = (const float*)d_in[1];
  const float* cw   = (const float*)d_in[2];
  const float* cb   = (const float*)d_in[3];
  const float* mw[3]  = {(const float*)d_in[4],  (const float*)d_in[8],  (const float*)d_in[12]};
  const float* mb[3]  = {(const float*)d_in[5],  (const float*)d_in[9],  (const float*)d_in[13]};
  const float* mg[3]  = {(const float*)d_in[6],  (const float*)d_in[10], (const float*)d_in[14]};
  const float* mbe[3] = {(const float*)d_in[7],  (const float*)d_in[11], (const float*)d_in[15]};
  const float* lrw  = (const float*)d_in[16];
  const float* lrb  = (const float*)d_in[17];
  const float* ew0  = (const float*)d_in[18];
  const float* eb0  = (const float*)d_in[19];
  const float* eg0  = (const float*)d_in[20];
  const float* ebe0 = (const float*)d_in[21];
  const float* ew1  = (const float*)d_in[22];
  const float* eb1  = (const float*)d_in[23];
  const float* eg1  = (const float*)d_in[24];
  const float* ebe1 = (const float*)d_in[25];
  const float* ew2  = (const float*)d_in[26];
  const float* eb2  = (const float*)d_in[27];
  const float* eg2  = (const float*)d_in[28];
  const float* ebe2 = (const float*)d_in[29];
  const float* gw   = (const float*)d_in[30];
  const float* gb   = (const float*)d_in[31];
  const float* tw1  = (const float*)d_in[32];
  const float* tb1  = (const float*)d_in[33];
  const float* tg   = (const float*)d_in[34];
  const float* tbe  = (const float*)d_in[35];
  const float* tw2  = (const float*)d_in[36];
  const float* tb2  = (const float*)d_in[37];
  float* out = (float*)d_out;

  // ---- workspace carve-up (256B aligned) ----
  char* base = (char*)d_ws;
  size_t off = 0;
  auto alloc = [&](size_t bytes) -> char* {
    char* p = base + off;
    off = (off + bytes + 255) & ~(size_t)255;
    return p;
  };
  bf16*  mw0bf  = (bf16*)alloc(256L * 512 * 2);
  bf16*  mw1bf  = (bf16*)alloc(512L * 256 * 2);
  bf16*  mw2bf  = (bf16*)alloc(256L * 128 * 2);
  bf16*  ew1bf  = (bf16*)alloc(13L * 512 * 512 * 2);
  bf16*  ew2bf  = (bf16*)alloc(13L * 512 * 256 * 2);
  bf16*  tw1bf  = (bf16*)alloc(6L * 256 * 256 * 2);
  float* ewsum0 = (float*)alloc(13L * 512 * 4);
  float* gwsum  = (float*)alloc(64L * 4);
  bf16*  exbf   = (bf16*)alloc((long)Bsz * 256 * 2);
  float* hbuf   = (float*)alloc((long)Bsz * 256 * 4);
  float* zbuf   = (float*)alloc((long)Bsz * 512 * 4);
  bf16*  abuf   = (bf16*)alloc((long)Bsz * 512 * 2);
  float* mbuf   = (float*)alloc((long)Bsz * 128 * 4);
  float* bnmean = (float*)alloc(512L * 4);
  float* bninv  = (float*)alloc(512L * 4);
  float* ybuf   = (float*)alloc((long)Bsz * 4);
  float* gcoef  = (float*)alloc(6L * Bsz * 10 * 4);
  bf16*  heA    = (bf16*)alloc(13L * BC * 512 * 2);
  bf16*  heB    = (bf16*)alloc(13L * BC * 512 * 2);
  bf16*  he3    = (bf16*)alloc(13L * BC * 256 * 2);
  bf16*  fusedb = (bf16*)alloc(6L * BC * 256 * 2);
  float* t1z    = (float*)alloc(6L * Bsz * 256 * 4);
  float* tmeanb = (float*)alloc(6L * 256 * 4);
  float* tinvb  = (float*)alloc(6L * 256 * 4);

  auto cvt = [&](const float* s, bf16* d, long n) {
    cvt_bf16_kernel<<<dim3((unsigned)((n + 255) / 256)), 256, 0, stream>>>(s, d, n);
  };

  // ---- weight down-conversion to bf16 ----
  cvt(mw[0], mw0bf, 256L * 512);
  cvt(mw[1], mw1bf, 512L * 256);
  cvt(mw[2], mw2bf, 256L * 128);
  cvt(ew1,   ew1bf, 13L * 512 * 512);
  cvt(ew2,   ew2bf, 13L * 512 * 256);
  cvt(tw1,   tw1bf, 6L * 256 * 256);

  // rank-1 collapses: column sums of ew0 / gw
  colsum_kernel<<<dim3((13 * 512 + 255) / 256), 256, 0, stream>>>(ew0, ewsum0, 13, 128, 512);
  colsum_kernel<<<dim3(1), 256, 0, stream>>>(gw, gwsum, 6, 128, 10);

  // ---- embedding + cross ----
  embed_cross_kernel<<<dim3(Bsz), 256, 0, stream>>>(x, emb, cw, cb, hbuf, exbf);

  // ---- MLP with batch-norm (full batch) ----
  const int mlpN[3] = {512, 256, 128};
  const int mlpK[3] = {256, 512, 256};
  const bf16* mwbf[3] = {mw0bf, mw1bf, mw2bf};
  const bf16* aIn = exbf;
  for (int i = 0; i < 3; ++i) {
    int N = mlpN[i], K = mlpK[i];
    gemm_bias_kernel<<<dim3(Bsz / 128, N / 64, 1), 256, 0, stream>>>(
        aIn, mwbf[i], mb[i], zbuf, Bsz, N, K, 0, 0, 0, 0);
    bn_stats_kernel<<<dim3(N), 256, 0, stream>>>(zbuf, Bsz, N, bnmean, bninv);
    long total = (long)Bsz * N;
    bn_relu_kernel<<<dim3((unsigned)((total + 255) / 256)), 256, 0, stream>>>(
        zbuf, bnmean, bninv, mg[i], mbe[i], (i < 2) ? abuf : nullptr,
        (i == 2) ? mbuf : nullptr, total, N);
    aIn = abuf;
  }

  // ---- logit y and gates ----
  y_kernel<<<dim3(Bsz), 256, 0, stream>>>(hbuf, mbuf, lrw, lrb, ybuf);
  gate_kernel<<<dim3(Bsz / 256), 256, 0, stream>>>(ybuf, gwsum, gb, gcoef);

  // ---- expert pipeline + fuse + tower GEMM, chunked over batch ----
  size_t lds512 = 32 * LPAD * 2 + 512 * LPAD * 2 + 32 * 512 * 4;
  size_t lds256 = 32 * LPAD * 2 + 256 * LPAD * 2 + 32 * 256 * 4;
  for (int c = 0; c < NCHK; ++c) {
    int b0 = c * BC;
    expert0_ln_kernel<<<dim3(BC, 13), 256, 0, stream>>>(ybuf, ewsum0, eb0, eg0, ebe0, heA, b0);
    gemm_ln_relu_kernel<512><<<dim3(BC / 32, 1, 13), 256, lds512, stream>>>(
        heA, ew1bf, eb1, eg1, ebe1, heB, 512,
        (long)BC * 512, 512L * 512, (long)BC * 512);
    gemm_ln_relu_kernel<256><<<dim3(BC / 32, 1, 13), 256, lds256, stream>>>(
        heB, ew2bf, eb2, eg2, ebe2, he3, 512,
        (long)BC * 512, 512L * 256, (long)BC * 256);
    fuse_kernel<<<dim3((unsigned)((6L * BC * 256 + 255) / 256)), 256, 0, stream>>>(
        gcoef, he3, x, fusedb, b0);
    gemm_bias_kernel<<<dim3(BC / 128, 256 / 64, 6), 256, 0, stream>>>(
        fusedb, tw1bf, tb1, t1z + (long)b0 * 256, BC, 256, 256,
        (long)BC * 256, 256L * 256, 256, (long)Bsz * 256);
  }

  // ---- tower batch-norm statistics + final selection ----
  for (int g = 0; g < 6; ++g)
    bn_stats_kernel<<<dim3(256), 256, 0, stream>>>(
        t1z + (long)g * Bsz * 256, Bsz, 256, tmeanb + g * 256, tinvb + g * 256);
  final_kernel<<<dim3(Bsz / 256), 256, 0, stream>>>(
      t1z, tmeanb, tinvb, tg, tbe, tw2, tb2, x, out);

  (void)in_sizes; (void)n_in; (void)out_size; (void)ws_size;
}
